// ImprovedPMDM_5368709120840
// MI455X (gfx1250) — matrix operand order repaired and checked by reading, not compiled, not run on hardware
//
#include <hip/hip_runtime.h>
#include <hip/hip_bf16.h>

// ---------------------------------------------------------------------------
// ImprovedPMDM forward for MI455X (gfx1250, wave32, WMMA 16x16x32 f16)
// ---------------------------------------------------------------------------

constexpr int HD    = 128;   // hidden dim
constexpr int NLIG  = 512;   // ligand atoms
constexpr int MPOC  = 1024;  // pocket atoms
constexpr int NTYPE = 10;

typedef __attribute__((ext_vector_type(16))) _Float16 v16h;
typedef __attribute__((ext_vector_type(8)))  float    v8f;

// SiLU with fast reciprocal: denominator >= 1, so v_rcp_f32 is safe and avoids
// the v_div_fixup/v_div_fmas precise-division expansion in the hot loops.
__device__ __forceinline__ float dev_silu(float x) {
  return x * __builtin_amdgcn_rcpf(1.0f + __expf(-x));
}

// K index inside a 16x32 f16 WMMA A-fragment for element e of `lane`.
// Lanes 0-15: halves 0..7 -> K 0..7, halves 8..15 -> K 16..23.
// Lanes 16-31: +8 on the K sub-range.
__device__ __forceinline__ int akidx(int e, int lane) {
  return (e & 7) | ((lane & 16) >> 1) | ((e & 8) << 1);
}

// B fragment (32x16 f16) from LDS weight tile stored TRANSPOSED as WT[n][k]
// (128x128). K values for a lane follow the A-fragment order (akidx).
__device__ __forceinline__ v16h load_bfragT(const _Float16* WT, int kb, int nb, int lane) {
  int n  = nb + (lane & 15);
  int k0 = kb;                     // K order per akidx (hi half-wave +8)
  const _Float16* src = WT + n * HD + k0;
  v16h b;
#pragma unroll
  for (int e = 0; e < 16; ++e) b[e] = src[akidx(e, lane)];
  return b;
}

// ---------------------------------------------------------------------------
// Generic fused GEMM: out[n,128] = act( X[n,128] @ W[128,128] + bias + addin )
// Compile-time specialized (no epilogue branches). n must be a multiple of 128.
// 8 waves, each wave does a 16x128 row tile: 4 k-chunks x 8 n-tiles WMMA.
// ---------------------------------------------------------------------------
template <int ACT, bool HAS_BIAS, bool HAS_ADDIN, bool OUT_HALF>
__global__ __launch_bounds__(256) void k_gemm128(
    const float* __restrict__ X, const float* __restrict__ W,
    const float* __restrict__ bias, const float* __restrict__ addin,
    float* __restrict__ outF, _Float16* __restrict__ outH, int n) {
  extern __shared__ char smem[];
  _Float16* sX = (_Float16*)smem;     // [128][128] row-major rows
  _Float16* sW = sX + HD * HD;        // [n][k] transposed weights
  int tid = threadIdx.x;
  int r0  = blockIdx.x * 128;
  for (int t = tid; t < HD * HD; t += 256) {
    int k = t & 127, nn = t >> 7;
    sW[t] = (_Float16)W[k * HD + nn];                    // transpose on fill
    sX[t] = (_Float16)X[(r0 + nn) * HD + k];
  }
  __syncthreads();

  int lane = tid & 31, wv = tid >> 5;
  int p = lane & 15, hi = (lane >> 4) & 1;
  int rowb = wv * 16;

  v16h a[4];
#pragma unroll
  for (int kc = 0; kc < 4; ++kc)
#pragma unroll
    for (int e = 0; e < 16; ++e)
      a[kc][e] = sX[(rowb + p) * HD + kc * 32 + akidx(e, lane)];

#pragma unroll
  for (int nt = 0; nt < 8; ++nt) {
    v8f acc = {0.f, 0.f, 0.f, 0.f, 0.f, 0.f, 0.f, 0.f};
#pragma unroll
    for (int kc = 0; kc < 4; ++kc)
      acc = __builtin_amdgcn_wmma_f32_16x16x32_f16(
          false, a[kc], false, load_bfragT(sW, kc * 32, nt * 16, lane),
          (short)0, acc, false, false);
    int col = nt * 16 + p;
    float bcol = HAS_BIAS ? bias[col] : 0.f;
#pragma unroll
    for (int v = 0; v < 8; ++v) {
      int row = r0 + rowb + v + hi * 8;
      float val = acc[v] + bcol;
      if (HAS_ADDIN) val += addin[row * HD + col];
      if (ACT)       val = dev_silu(val);
      if (OUT_HALF)  outH[row * HD + col] = (_Float16)val;
      else           outF[row * HD + col] = val;
    }
  }
  (void)n;
}

// ---------------------------------------------------------------------------
// Fused local EGNN pair kernel.  Block = 16 ligand i's x 128 j's.
// Per wave task (i, 16 j's):  pre -> silu -> @msg2 (WMMA) -> mask diag
//   -> rowsum into agg  and  @c1 -> silu -> .c2 -> scalar -> coord delta.
// msgs (N^2 x H, 134 MB) never touches HBM.
// ---------------------------------------------------------------------------
__global__ __launch_bounds__(256) void k_pairs_local(
    const _Float16* __restrict__ Af, const _Float16* __restrict__ Bf,
    const float* __restrict__ coords,
    const float* __restrict__ distw, const float* __restrict__ b1,
    const float* __restrict__ W2,   const float* __restrict__ b2,
    const float* __restrict__ C1,   const float* __restrict__ c1b,
    const float* __restrict__ c2w,  const float* __restrict__ c2b,
    float* __restrict__ agg, float* __restrict__ cdelta) {
  extern __shared__ char smem[];
  _Float16* sW2  = (_Float16*)smem;        // 128x128 transposed [n][k]
  _Float16* sC1  = sW2 + HD * HD;          // 128x128 transposed [n][k]
  _Float16* sB   = sC1 + HD * HD;          // 128 j rows x 128
  _Float16* sA   = sB + 128 * HD;          // 16 i rows x 128
  _Float16* sMsg = sA + 16 * HD;           // 8 waves x 16x128 staging
  float* sAgg  = (float*)(sMsg + 8 * 16 * HD); // 16x128
  float* sV    = sAgg + 16 * HD;           // distw|b1|b2|c1b|c2w  (5x128)
  float* sCi   = sV + 5 * HD;              // 16x3
  float* sCj   = sCi + 48;                 // 128x3
  float* sCd   = sCj + 384;                // 16x3
  float* sScal = sCd + 48;                 // 8 waves x 16

  int tid = threadIdx.x;
  int iblock = blockIdx.x >> 2;            // 32 i-blocks
  int jc     = blockIdx.x & 3;             // 4 j-chunks of 128
  int i0 = iblock * 16, j0 = jc * 128;

  for (int t = tid; t < HD * HD; t += 256) {
    int k = t & 127, nn = t >> 7;
    sW2[t] = (_Float16)W2[k * HD + nn];
    sC1[t] = (_Float16)C1[k * HD + nn];
    sB[t]  = Bf[(j0 + nn) * HD + k];
  }
  for (int t = tid; t < 16 * HD; t += 256) {
    sA[t]   = Af[(i0 + (t >> 7)) * HD + (t & 127)];
    sAgg[t] = 0.f;
  }
  if (tid < HD) {
    sV[tid]          = distw[tid];
    sV[HD + tid]     = b1[tid];
    sV[2 * HD + tid] = b2[tid];
    sV[3 * HD + tid] = c1b[tid];
    sV[4 * HD + tid] = c2w[tid];
  }
  if (tid < 48) { sCi[tid] = coords[i0 * 3 + tid]; sCd[tid] = 0.f; }
  for (int t = tid; t < 384; t += 256) sCj[t] = coords[j0 * 3 + t];
  __syncthreads();

  int lane = tid & 31, wv = tid >> 5;
  int p = lane & 15, hi = (lane >> 4) & 1;
  _Float16* msg = sMsg + wv * (16 * HD);
  float c2bb = c2b[0];

  for (int task = wv; task < 128; task += 8) {   // 16 i x 8 j-tiles
    int il = task >> 3, jt = task & 7;
    int jl = jt * 16 + p;
    int i  = i0 + il;
    float rx = sCi[il * 3 + 0] - sCj[jl * 3 + 0];
    float ry = sCi[il * 3 + 1] - sCj[jl * 3 + 1];
    float rz = sCi[il * 3 + 2] - sCj[jl * 3 + 2];
    float sq = rx * rx + ry * ry + rz * rz;
    float dist = (sq == 0.f) ? 0.f : sqrtf(sq);

    // silu(pre) A-fragments
    v16h a[4];
#pragma unroll
    for (int kc = 0; kc < 4; ++kc)
#pragma unroll
      for (int e = 0; e < 16; ++e) {
        int k = kc * 32 + akidx(e, lane);
        float pre = (float)sA[il * HD + k] + (float)sB[jl * HD + k]
                  + dist * sV[k] + sV[HD + k];
        a[kc][e] = (_Float16)dev_silu(pre);
      }

    // msgs = silu(pre) @ W2 + b2 ; mask diagonal ; stage + accumulate agg
#pragma unroll
    for (int nt = 0; nt < 8; ++nt) {
      v8f acc = {0.f, 0.f, 0.f, 0.f, 0.f, 0.f, 0.f, 0.f};
#pragma unroll
      for (int kc = 0; kc < 4; ++kc)
        acc = __builtin_amdgcn_wmma_f32_16x16x32_f16(
            false, a[kc], false, load_bfragT(sW2, kc * 32, nt * 16, lane),
            (short)0, acc, false, false);
      int col = nt * 16 + p;
      float bcol = sV[2 * HD + col];
      float part = 0.f;
#pragma unroll
      for (int v = 0; v < 8; ++v) {
        int row = v + hi * 8;
        float m = acc[v] + bcol;
        if (j0 + jt * 16 + row == i) m = 0.f;     // ~eye mask
        msg[row * HD + col] = (_Float16)m;
        part += m;
      }
      atomicAdd(&sAgg[il * HD + col], part);
    }

    if (lane < 16) sScal[wv * 16 + lane] = 0.f;

    // scalar = (silu(msgs @ C1 + c1b)) . c2w + c2b
    v16h a2[4];
#pragma unroll
    for (int kc = 0; kc < 4; ++kc)
#pragma unroll
      for (int e = 0; e < 16; ++e)
        a2[kc][e] = msg[p * HD + kc * 32 + akidx(e, lane)];
    float sp[8] = {0.f, 0.f, 0.f, 0.f, 0.f, 0.f, 0.f, 0.f};
#pragma unroll
    for (int nt = 0; nt < 8; ++nt) {
      v8f acc = {0.f, 0.f, 0.f, 0.f, 0.f, 0.f, 0.f, 0.f};
#pragma unroll
      for (int kc = 0; kc < 4; ++kc)
        acc = __builtin_amdgcn_wmma_f32_16x16x32_f16(
            false, a2[kc], false, load_bfragT(sC1, kc * 32, nt * 16, lane),
            (short)0, acc, false, false);
      int col = nt * 16 + p;
      float cb = sV[3 * HD + col], cw = sV[4 * HD + col];
#pragma unroll
      for (int v = 0; v < 8; ++v) sp[v] += dev_silu(acc[v] + cb) * cw;
    }
#pragma unroll
    for (int v = 0; v < 8; ++v) atomicAdd(&sScal[wv * 16 + v + hi * 8], sp[v]);

    if (lane < 16) {
      float s = sScal[wv * 16 + p] + c2bb;
      atomicAdd(&sCd[il * 3 + 0], s * rx);
      atomicAdd(&sCd[il * 3 + 1], s * ry);
      atomicAdd(&sCd[il * 3 + 2], s * rz);
    }
  }
  __syncthreads();
  for (int t = tid; t < 16 * HD; t += 256)
    atomicAdd(&agg[(i0 + (t >> 7)) * HD + (t & 127)], sAgg[t]);
  if (tid < 48) atomicAdd(&cdelta[i0 * 3 + tid], sCd[tid]);
}

// ---------------------------------------------------------------------------
// Fused global (ligand x pocket) pair kernel: msgs sum only.
// ---------------------------------------------------------------------------
__global__ __launch_bounds__(256) void k_pairs_global(
    const _Float16* __restrict__ Ag, const _Float16* __restrict__ Bg,
    const float* __restrict__ ligc, const float* __restrict__ pocc,
    const float* __restrict__ distw, const float* __restrict__ b1,
    const float* __restrict__ W2, const float* __restrict__ b2,
    float* __restrict__ agg) {
  extern __shared__ char smem[];
  _Float16* sW2 = (_Float16*)smem;          // transposed [n][k]
  _Float16* sB  = sW2 + HD * HD;            // 128 pocket rows
  _Float16* sA  = sB + 128 * HD;            // 16 ligand rows
  float* sAgg = (float*)(sA + 16 * HD);     // 16x128
  float* sV   = sAgg + 16 * HD;             // distw|b1|b2
  float* sCi  = sV + 3 * HD;                // 16x3
  float* sCj  = sCi + 48;                   // 128x3

  int tid = threadIdx.x;
  int iblock = blockIdx.x >> 3;             // 32 i-blocks
  int jc     = blockIdx.x & 7;              // 8 j-chunks of 128 (M=1024)
  int i0 = iblock * 16, j0 = jc * 128;

  for (int t = tid; t < HD * HD; t += 256) {
    int k = t & 127, nn = t >> 7;
    sW2[t] = (_Float16)W2[k * HD + nn];
    sB[t]  = Bg[(j0 + nn) * HD + k];
  }
  for (int t = tid; t < 16 * HD; t += 256) {
    sA[t]   = Ag[(i0 + (t >> 7)) * HD + (t & 127)];
    sAgg[t] = 0.f;
  }
  if (tid < HD) {
    sV[tid]          = distw[tid];
    sV[HD + tid]     = b1[tid];
    sV[2 * HD + tid] = b2[tid];
  }
  if (tid < 48) sCi[tid] = ligc[i0 * 3 + tid];
  for (int t = tid; t < 384; t += 256) sCj[t] = pocc[j0 * 3 + t];
  __syncthreads();

  int lane = tid & 31, wv = tid >> 5;
  int p = lane & 15, hi = (lane >> 4) & 1;

  for (int task = wv; task < 128; task += 8) {
    int il = task >> 3, jt = task & 7;
    int jl = jt * 16 + p;
    float rx = sCi[il * 3 + 0] - sCj[jl * 3 + 0];
    float ry = sCi[il * 3 + 1] - sCj[jl * 3 + 1];
    float rz = sCi[il * 3 + 2] - sCj[jl * 3 + 2];
    float sq = rx * rx + ry * ry + rz * rz;
    float dist = (sq == 0.f) ? 0.f : sqrtf(sq);

    v16h a[4];
#pragma unroll
    for (int kc = 0; kc < 4; ++kc)
#pragma unroll
      for (int e = 0; e < 16; ++e) {
        int k = kc * 32 + akidx(e, lane);
        float pre = (float)sA[il * HD + k] + (float)sB[jl * HD + k]
                  + dist * sV[k] + sV[HD + k];
        a[kc][e] = (_Float16)dev_silu(pre);
      }
#pragma unroll
    for (int nt = 0; nt < 8; ++nt) {
      v8f acc = {0.f, 0.f, 0.f, 0.f, 0.f, 0.f, 0.f, 0.f};
#pragma unroll
      for (int kc = 0; kc < 4; ++kc)
        acc = __builtin_amdgcn_wmma_f32_16x16x32_f16(
            false, a[kc], false, load_bfragT(sW2, kc * 32, nt * 16, lane),
            (short)0, acc, false, false);
      int col = nt * 16 + p;
      float bcol = sV[2 * HD + col];
      float part = 0.f;
#pragma unroll
      for (int v = 0; v < 8; ++v) part += acc[v] + bcol;
      atomicAdd(&sAgg[il * HD + col], part);
    }
  }
  __syncthreads();
  for (int t = tid; t < 16 * HD; t += 256)
    atomicAdd(&agg[(i0 + (t >> 7)) * HD + (t & 127)], sAgg[t]);
}

// ---------------------------------------------------------------------------
// Small VALU kernels
// ---------------------------------------------------------------------------
__global__ __launch_bounds__(128) void k_temb(
    const float* __restrict__ t, const float* __restrict__ t1W,
    const float* __restrict__ t1b, const float* __restrict__ t2W,
    const float* __restrict__ t2b, float* __restrict__ temb) {
  __shared__ float emb[HD], h1[HD];
  int d = threadIdx.x;
  float tv = t[0];
  if (d < 64) {
    float f = __expf(-logf(10000.f) * (float)d / 64.f);
    emb[d]      = __sinf(tv * f);
    emb[d + 64] = __cosf(tv * f);
  }
  __syncthreads();
  float s = t1b[d];
  for (int k = 0; k < HD; ++k) s += emb[k] * t1W[k * HD + d];
  h1[d] = fmaxf(s, 0.f);
  __syncthreads();
  float o = t2b[d];
  for (int k = 0; k < HD; ++k) o += h1[k] * t2W[k * HD + d];
  temb[d] = o;
}

__global__ __launch_bounds__(128) void k_embed_lig(
    const int* __restrict__ atoms, const float* __restrict__ emb,
    const float* __restrict__ noisy, const float* __restrict__ teW,
    const float* __restrict__ teb, float* __restrict__ out) {
  int i = blockIdx.x, d = threadIdx.x;
  float v = emb[atoms[i] * HD + d] + teb[d];
#pragma unroll
  for (int t = 0; t < NTYPE; ++t) v += noisy[i * NTYPE + t] * teW[t * HD + d];
  out[i * HD + d] = v;
}

__global__ __launch_bounds__(128) void k_embed_poc(
    const int* __restrict__ atoms, const float* __restrict__ emb,
    float* __restrict__ out) {
  int i = blockIdx.x, d = threadIdx.x;
  out[i * HD + d] = emb[atoms[i] * HD + d];
}

__global__ __launch_bounds__(256) void k_pocmean(const float* __restrict__ pc,
                                                 float* __restrict__ pm) {
  __shared__ float acc[3];
  if (threadIdx.x < 3) acc[threadIdx.x] = 0.f;
  __syncthreads();
  float s0 = 0.f, s1 = 0.f, s2 = 0.f;
  for (int i = threadIdx.x; i < MPOC; i += 256) {
    s0 += pc[i * 3 + 0]; s1 += pc[i * 3 + 1]; s2 += pc[i * 3 + 2];
  }
  atomicAdd(&acc[0], s0); atomicAdd(&acc[1], s1); atomicAdd(&acc[2], s2);
  __syncthreads();
  if (threadIdx.x < 3) pm[threadIdx.x] = acc[threadIdx.x] * (1.f / (float)MPOC);
}

__global__ __launch_bounds__(128) void k_vecbias(
    const float* __restrict__ temb, const float* __restrict__ W,
    const float* __restrict__ b, float* __restrict__ out) {
  __shared__ float te[HD];
  int d = threadIdx.x;
  te[d] = temb[d];
  __syncthreads();
  float s = b[d];
  for (int k = 0; k < HD; ++k) s += te[k] * W[k * HD + d];
  out[d] = s;
}

__global__ __launch_bounds__(256) void k_axpby(
    float* __restrict__ out, const float* __restrict__ a,
    const float* __restrict__ b, float sa, float sb, int n) {
  int i = blockIdx.x * blockDim.x + threadIdx.x;
  if (i < n) out[i] = sa * a[i] + sb * b[i];
}

__global__ __launch_bounds__(128) void k_dotc2(
    const float* __restrict__ X, const float* __restrict__ wvec,
    const float* __restrict__ b, float* __restrict__ s) {
  __shared__ float red[HD];
  int i = blockIdx.x, d = threadIdx.x;
  red[d] = X[i * HD + d] * wvec[d];
  __syncthreads();
  for (int off = 64; off > 0; off >>= 1) {
    if (d < off) red[d] += red[d + off];
    __syncthreads();
  }
  if (d == 0) s[i] = red[0] + b[0];
}

__global__ __launch_bounds__(32) void k_coord_final(
    const float* __restrict__ cur, const float* __restrict__ cd,
    const float* __restrict__ s, const float* __restrict__ pm,
    float* __restrict__ nxt) {
  int i = blockIdx.x, d = threadIdx.x;
  if (d < 3) {
    float c = cur[i * 3 + d];
    nxt[i * 3 + d] = c + 0.5f * cd[i * 3 + d] + 0.5f * s[i] * (c - pm[d]);
  }
}

__global__ __launch_bounds__(32) void k_heads(
    const float* __restrict__ feat, const float* __restrict__ cW,
    const float* __restrict__ cb, const float* __restrict__ tW,
    const float* __restrict__ tb, float* __restrict__ out) {
  int i = blockIdx.x, o = threadIdx.x;
  if (o < 3) {
    float s = cb[o];
    for (int k = 0; k < HD; ++k) s += feat[i * HD + k] * cW[k * 3 + o];
    out[i * 3 + o] = s;
  } else if (o < 13) {
    int oo = o - 3;
    float s = tb[oo];
    for (int k = 0; k < HD; ++k) s += feat[i * HD + k] * tW[k * NTYPE + oo];
    out[NLIG * 3 + i * NTYPE + oo] = s;
  }
}

// ---------------------------------------------------------------------------
// Host orchestration
// ---------------------------------------------------------------------------
static constexpr size_t GEMM_LDS   = 2ull * HD * HD * sizeof(_Float16);              // 64 KB
static constexpr size_t LOCAL_LDS  = (size_t)(HD*HD + HD*HD + 128*HD + 16*HD + 8*16*HD) * 2
                                   + (size_t)(16*HD + 5*HD + 48 + 384 + 48 + 128) * 4; // ~145 KB
static constexpr size_t GLOBAL_LDS = (size_t)(HD*HD + 128*HD + 16*HD) * 2
                                   + (size_t)(16*HD + 3*HD + 48 + 384) * 4;            // ~79 KB

extern "C" void kernel_launch(void* const* d_in, const int* in_sizes, int n_in,
                              void* d_out, int out_size, void* d_ws, size_t ws_size,
                              hipStream_t stream) {
  (void)in_sizes; (void)n_in; (void)out_size; (void)ws_size;

  // ---- inputs (setup_inputs order; params pytree leaves alphabetical) ----
  const int*   lig_atoms = (const int*)d_in[0];
  const float* lig_coords = (const float*)d_in[1];
  const int*   poc_atoms = (const int*)d_in[2];
  const float* poc_coords = (const float*)d_in[3];
  const float* noisy = (const float*)d_in[4];
  const float* tval  = (const float*)d_in[5];
  auto P = [&](int i) { return (const float*)d_in[i]; };
  // 6: atom_embed ; 7/8: coord_head W,b ; 9..20: fusion[i] f1.W f1.b f2.W f2.b
  // 21..56: global[i] (c1W c1b c2W c2b f1W f1b f2W f2b m1W m1b m2W m2b)
  // 57..92: local[i] same ; 93..94: t1 ; 95..96: t2 ; 97/98: type_embed ; 99/100: type_head
  const float* atom_embed = P(6);

  // ---- workspace carve ----
  char* wp = (char*)d_ws;
  auto carve = [&](size_t bytes) { char* q = wp; wp += (bytes + 255) & ~size_t(255); return q; };
  float* ligF0 = (float*)carve(NLIG * HD * 4);
  float* ligF1 = (float*)carve(NLIG * HD * 4);
  float* pocF  = (float*)carve(MPOC * HD * 4);
  float* temb  = (float*)carve(HD * 4);
  float* curC  = (float*)carve(NLIG * 3 * 4);
  float* newC  = (float*)carve(NLIG * 3 * 4);
  float* aggL  = (float*)carve(NLIG * HD * 4);
  float* aggG  = (float*)carve(NLIG * HD * 4);
  float* cdL   = (float*)carve(NLIG * 3 * 4);
  float* Xbuf  = (float*)carve(NLIG * HD * 4);
  float* hid   = (float*)carve(NLIG * HD * 4);
  float* lf    = (float*)carve(NLIG * HD * 4);
  float* gf    = (float*)carve(NLIG * HD * 4);
  float* fbuf  = (float*)carve(NLIG * HD * 4);
  float* aggm  = (float*)carve(NLIG * HD * 4);
  float* c1h   = (float*)carve(NLIG * HD * 4);
  float* svec  = (float*)carve(NLIG * 4);
  float* pmean = (float*)carve(3 * 4);
  float* vb    = (float*)carve(HD * 4);
  _Float16* Af = (_Float16*)carve(NLIG * HD * 2);
  _Float16* Bf = (_Float16*)carve(NLIG * HD * 2);
  _Float16* Ag = (_Float16*)carve(NLIG * HD * 2);
  _Float16* Bg = (_Float16*)carve(MPOC * HD * 2);

  // specialized GEMM dispatch (n always a multiple of 128 here)
  auto gemmH = [&](const float* X, const float* W, _Float16* oH, int n) {
    k_gemm128<0, false, false, true><<<n / 128, 256, GEMM_LDS, stream>>>(
        X, W, nullptr, nullptr, nullptr, oH, n);
  };
  auto gemmF = [&](const float* X, const float* W, float* oF, int n) {
    k_gemm128<0, false, false, false><<<n / 128, 256, GEMM_LDS, stream>>>(
        X, W, nullptr, nullptr, oF, nullptr, n);
  };
  auto gemmB = [&](const float* X, const float* W, const float* bias, float* oF, int n) {
    k_gemm128<0, true, false, false><<<n / 128, 256, GEMM_LDS, stream>>>(
        X, W, bias, nullptr, oF, nullptr, n);
  };
  auto gemmBS = [&](const float* X, const float* W, const float* bias, float* oF, int n) {
    k_gemm128<1, true, false, false><<<n / 128, 256, GEMM_LDS, stream>>>(
        X, W, bias, nullptr, oF, nullptr, n);
  };
  auto gemmBAS = [&](const float* X, const float* W, const float* bias,
                     const float* addin, float* oF, int n) {
    k_gemm128<1, true, true, false><<<n / 128, 256, GEMM_LDS, stream>>>(
        X, W, bias, addin, oF, nullptr, n);
  };

  // ---- prologue ----
  k_temb<<<1, 128, 0, stream>>>(tval, P(93), P(94), P(95), P(96), temb);
  k_embed_poc<<<MPOC, 128, 0, stream>>>(poc_atoms, atom_embed, pocF);
  k_embed_lig<<<NLIG, 128, 0, stream>>>(lig_atoms, atom_embed, noisy, P(97), P(98), ligF0);
  k_pocmean<<<1, 256, 0, stream>>>(poc_coords, pmean);
  k_axpby<<<(NLIG * 3 + 255) / 256, 256, 0, stream>>>(curC, lig_coords, lig_coords, 1.f, 0.f, NLIG * 3);

  float* featCur = ligF0; float* featNxt = ligF1;
  float* cCur = curC;     float* cNxt = newC;

  for (int l = 0; l < 3; ++l) {
    const int fb = 9 + l * 4, gb = 21 + l * 12, lb = 57 + l * 12;
    const float *Lc1W = P(lb+0), *Lc1b = P(lb+1), *Lc2W = P(lb+2), *Lc2b = P(lb+3);
    const float *Lf1W = P(lb+4), *Lf1b = P(lb+5), *Lf2W = P(lb+6), *Lf2b = P(lb+7);
    const float *Lm1W = P(lb+8), *Lm1b = P(lb+9), *Lm2W = P(lb+10), *Lm2b = P(lb+11);
    const float *Gc1W = P(gb+0), *Gc1b = P(gb+1), *Gc2W = P(gb+2), *Gc2b = P(gb+3);
    const float *Gf1W = P(gb+4), *Gf1b = P(gb+5), *Gf2W = P(gb+6), *Gf2b = P(gb+7);
    const float *Gm1W = P(gb+8), *Gm1b = P(gb+9), *Gm2W = P(gb+10), *Gm2b = P(gb+11);
    const float *F1W = P(fb+0), *F1b = P(fb+1), *F2W = P(fb+2), *F2b = P(fb+3);

    // msg1 projections (A = feat @ W1[:H], B = feat @ W1[H:2H]) in f16
    gemmH(featCur, Lm1W,           Af, NLIG);
    gemmH(featCur, Lm1W + HD * HD, Bf, NLIG);
    gemmH(featCur, Gm1W,           Ag, NLIG);
    gemmH(pocF,    Gm1W + HD * HD, Bg, MPOC);

    hipMemsetAsync(aggL, 0, NLIG * HD * 4, stream);
    hipMemsetAsync(aggG, 0, NLIG * HD * 4, stream);
    hipMemsetAsync(cdL,  0, NLIG * 3 * 4, stream);

    // fused pairwise message passing (the ~35 GFLOP/layer core)
    k_pairs_local<<<128, 256, LOCAL_LDS, stream>>>(
        Af, Bf, cCur, Lm1W + 2 * HD * HD, Lm1b, Lm2W, Lm2b,
        Lc1W, Lc1b, Lc2W, Lc2b, aggL, cdL);
    k_pairs_global<<<256, 256, GLOBAL_LDS, stream>>>(
        Ag, Bg, cCur, poc_coords, Gm1W + 2 * HD * HD, Gm1b, Gm2W, Gm2b, aggG);

    // local node update: lf = feat2( silu((feat+agg)@f1[:H] + temb@f1[H:] + b) )
    k_vecbias<<<1, 128, 0, stream>>>(temb, Lf1W + HD * HD, Lf1b, vb);
    k_axpby<<<(NLIG * HD + 255) / 256, 256, 0, stream>>>(Xbuf, featCur, aggL, 1.f, 1.f, NLIG * HD);
    gemmBS(Xbuf, Lf1W, vb, hid, NLIG);
    gemmB (hid,  Lf2W, Lf2b, lf, NLIG);

    // global node update (agg is mean over M)
    k_vecbias<<<1, 128, 0, stream>>>(temb, Gf1W + HD * HD, Gf1b, vb);
    k_axpby<<<(NLIG * HD + 255) / 256, 256, 0, stream>>>(Xbuf, featCur, aggG, 1.f, 1.f / (float)MPOC, NLIG * HD);
    gemmBS(Xbuf, Gf1W, vb, hid, NLIG);
    gemmB (hid,  Gf2W, Gf2b, gf, NLIG);

    // global coord scalar: s = c2( silu( c1(mean_agg) ) )
    k_axpby<<<(NLIG * HD + 255) / 256, 256, 0, stream>>>(aggm, aggG, aggG, 1.f / (float)MPOC, 0.f, NLIG * HD);
    gemmBS(aggm, Gc1W, Gc1b, c1h, NLIG);
    k_dotc2<<<NLIG, 128, 0, stream>>>(c1h, Gc2W, Gc2b, svec);

    // fusion: feat' = f2( silu(lf@f1[:H] + gf@f1[H:] + b) )
    gemmF  (lf, F1W, fbuf, NLIG);
    gemmBAS(gf, F1W + HD * HD, F1b, fbuf, hid, NLIG);
    gemmB  (hid, F2W, F2b, featNxt, NLIG);

    // coords' = (lc + gc)/2 = cur + 0.5*cdL + 0.5*s*(cur - poc_mean)
    k_coord_final<<<NLIG, 32, 0, stream>>>(cCur, cdL, svec, pmean, cNxt);

    { float* t0 = featCur; featCur = featNxt; featNxt = t0; }
    { float* t1 = cCur;    cCur = cNxt;       cNxt = t1;    }
  }

  // ---- heads: coord (512x3) then type (512x10), concatenated flat ----
  k_heads<<<NLIG, 32, 0, stream>>>(featCur, P(7), P(8), P(99), P(100), (float*)d_out);
}
